// AutoAttention_deit_34170759807639
// MI455X (gfx1250) — compile-verified, matrix operand order checked by reading, not run on hardware
//
#include <hip/hip_runtime.h>
#include <hip/hip_bf16.h>
#include <cstdint>

typedef __attribute__((ext_vector_type(16))) _Float16 v16h;
typedef __attribute__((ext_vector_type(8)))  _Float16 v8h;
typedef __attribute__((ext_vector_type(8)))  float    v8f;

#define NB   32
#define LTOK 577
#define CDIM 768
#define NHD  12
#define HD   64
#define MTOT (NB * LTOK)      // 18464, divisible by 16
#define C3   (3 * CDIM)       // 2304
#define LPAD 608              // 19 * 32 (key blocks, zero padded)
#define ATT_SCALE 0.125f      // 64^-0.5
#define MTILES (MTOT / 16)    // 1154
#define MBLKS  ((MTILES + 7) / 8)  // 145 blocks of 8 wave-tiles

// ---------------------------------------------------------------------------
// WMMA fragment loaders. A-matrix 16x32 f16 layout (ISA 7.12.2):
// lane&15 selects row; lanes<16 hold K = {0..7, 16..23}, lanes>=16 hold
// K = {8..15, 24..31}. Each half is one contiguous 16-byte load.
// B fragments from a row-major [N][K] array (B = W^T) load identically.
// ---------------------------------------------------------------------------
__device__ __forceinline__ v16h frag_rm(const _Float16* p, int ld, int lane) {
  const _Float16* q = p + (size_t)(lane & 15) * ld + ((lane >> 4) << 3);
  union { v16h v; v8h h[2]; } u;
  u.h[0] = *(const v8h*)q;
  u.h[1] = *(const v8h*)(q + 16);
  return u.v;
}

__device__ __forceinline__ v16h frag_rm_clamp(const _Float16* base, int row0, int maxrow,
                                              int ld, int kofs, int lane) {
  int r = row0 + (lane & 15);
  if (r > maxrow) r = maxrow;
  const _Float16* q = base + (size_t)r * ld + kofs + ((lane >> 4) << 3);
  union { v16h v; v8h h[2]; } u;
  u.h[0] = *(const v8h*)q;
  u.h[1] = *(const v8h*)(q + 16);
  return u.v;
}

__device__ __forceinline__ v8f wmma_f16(v16h a, v16h b, v8f c) {
  return __builtin_amdgcn_wmma_f32_16x16x32_f16(false, a, false, b, (short)0, c,
                                                false, false);
}

// ---------------------------------------------------------------------------
// Async Global -> LDS copy of one 64x32 f16 B tile (4 KB), 256 threads x 16B.
// Emits GLOBAL_LOAD_ASYNC_TO_LDS_B128 (ASYNCcnt); LDS offset = low 32 bits of
// the flat shared pointer.
// ---------------------------------------------------------------------------
__device__ __forceinline__ void async_copy_btile(_Float16* dstlds,
                                                 const _Float16* src, int tid) {
  int n = tid >> 2;                      // 0..63 : B row (output column)
  int c = tid & 3;                       // 0..3  : 8-half chunk within row
  const _Float16* g = src + (size_t)n * CDIM + (c << 3);
  uint32_t lds = (uint32_t)(size_t)(dstlds + n * 32 + (c << 3));
  asm volatile("global_load_async_to_lds_b128 %0, %1, off"
               :: "v"(lds), "v"(g) : "memory");
}

__device__ __forceinline__ void wait_asynccnt0() {
  asm volatile("s_wait_asynccnt 0x0" ::: "memory");
}

// ---------------------------------------------------------------------------
// One K-step of the block GEMM: acc[4] += A(16x32) @ Blds(32x64)
// ---------------------------------------------------------------------------
__device__ __forceinline__ void gemm_compute_step(const _Float16* __restrict__ a0,
                                                  int k, const _Float16* bl,
                                                  int lane, v8f acc[4]) {
  v16h af = frag_rm(a0 + k, CDIM, lane);
#pragma unroll
  for (int t = 0; t < 4; ++t) {
    v16h bf = frag_rm(bl + t * 16 * 32, 32, lane);  // ds_load_b128 pairs
    acc[t] = wmma_f16(af, bf, acc[t]);
  }
}

// ---------------------------------------------------------------------------
// Shared GEMM mainloop: B staged in LDS, double-buffered async copies with
// compile-time buffer addresses (unroll-by-2 so LDS offsets are immediates).
// All 256 threads of the block must call this (barriers inside).
// ---------------------------------------------------------------------------
__device__ __forceinline__ void gemm_mainloop(const _Float16* __restrict__ a0,
                                              const _Float16* __restrict__ bglob,
                                              _Float16* bsmem,   // [2][64*32]
                                              int tid, int lane, v8f acc[4]) {
  _Float16* buf0 = bsmem;
  _Float16* buf1 = bsmem + 64 * 32;
  const int NSTEP = CDIM / 32;           // 24, even
  async_copy_btile(buf0, bglob, tid);
#pragma unroll 1
  for (int kt = 0; kt < NSTEP; kt += 2) {
    wait_asynccnt0();
    __syncthreads();                     // tile kt visible to all waves
    if (kt + 1 < NSTEP) async_copy_btile(buf1, bglob + (kt + 1) * 32, tid);
    __builtin_prefetch(a0 + kt * 32 + 128, 0, 1);   // global_prefetch_b8
    gemm_compute_step(a0, kt * 32, buf0, lane, acc);
    __syncthreads();                     // done reading buf0

    wait_asynccnt0();
    __syncthreads();                     // tile kt+1 visible
    if (kt + 2 < NSTEP) async_copy_btile(buf0, bglob + (kt + 2) * 32, tid);
    gemm_compute_step(a0, (kt + 1) * 32, buf1, lane, acc);
    __syncthreads();                     // done reading buf1
  }
}

// ---------------------------------------------------------------------------
// f32 -> f16 convert
// ---------------------------------------------------------------------------
__global__ void f32_to_f16_kernel(const float* __restrict__ in,
                                  _Float16* __restrict__ out, size_t n) {
  size_t i = (size_t)blockIdx.x * blockDim.x + threadIdx.x;
  if (i < n) out[i] = (_Float16)in[i];
}

// ---------------------------------------------------------------------------
// QKV GEMM: [MTOT x 768] @ [2304 x 768]^T -> scatter f16 qkv[which][n][h][l][d]
// Block = 8 waves = 128x64 tile; B tile shared in LDS (async double buffer).
// ---------------------------------------------------------------------------
__global__ __launch_bounds__(256)
void qkv_gemm_kernel(const _Float16* __restrict__ xh, const _Float16* __restrict__ wh,
                     _Float16* __restrict__ qkv) {
  __shared__ __align__(16) _Float16 bsmem[2][64 * 32];
  const int lane = threadIdx.x & 31;
  const int wv = threadIdx.x >> 5;
  int nt = blockIdx.x / MBLKS;
  int mb = blockIdx.x - nt * MBLKS;
  int mt = mb * 8 + wv;
  bool mvalid = mt < MTILES;
  int mtc = mvalid ? mt : MTILES - 1;

  const _Float16* a0 = xh + (size_t)mtc * 16 * CDIM;
  const _Float16* b0 = wh + (size_t)nt * 64 * CDIM;

  v8f acc[4] = {};
  gemm_mainloop(a0, b0, &bsmem[0][0], threadIdx.x, lane, acc);
  if (!mvalid) return;

  // D layout: VGPR r -> row mt*16 + r + 8*(lane>=16), col = tile_col + (lane&15)
#pragma unroll
  for (int t = 0; t < 4; ++t) {
    int col = nt * 64 + t * 16 + (lane & 15);       // 0..2303 = (which, head, d)
    int which = col / CDIM;
    int rem = col - which * CDIM;
    int head = rem >> 6;
    int d = rem & 63;
#pragma unroll
    for (int r = 0; r < 8; ++r) {
      int row = mt * 16 + r + ((lane >> 4) << 3);   // token = n*L + l
      int nb = row / LTOK;
      int l = row - nb * LTOK;
      size_t idx = ((((size_t)which * NB + nb) * NHD + head) * LTOK + l) * HD + d;
      qkv[idx] = (_Float16)acc[t][r];
    }
  }
}

// ---------------------------------------------------------------------------
// Transpose V per head: vt[nh][d][lp] = v[nh][lp][d], zero-padded to LPAD keys
// ---------------------------------------------------------------------------
__global__ void vt_kernel(const _Float16* __restrict__ qkv, _Float16* __restrict__ vt) {
  size_t idx = (size_t)blockIdx.x * blockDim.x + threadIdx.x;
  const size_t total = (size_t)NB * NHD * HD * LPAD;
  if (idx >= total) return;
  int lp = (int)(idx % LPAD);
  int d = (int)((idx / LPAD) % HD);
  size_t nh = idx / ((size_t)LPAD * HD);
  _Float16 val = (_Float16)0.f;
  if (lp < LTOK)
    val = qkv[(((size_t)2 * NB * NHD + nh) * LTOK + lp) * HD + d];
  vt[idx] = val;
}

// ---------------------------------------------------------------------------
// Flash attention: one (n,h) head + 128-query band per block; each wave owns
// 16 query rows. Key blocks of 32; online softmax with half-wave shuffles.
// ---------------------------------------------------------------------------
__global__ __launch_bounds__(256)
void attn_kernel(const _Float16* __restrict__ qkv, const _Float16* __restrict__ vt,
                 _Float16* __restrict__ ob) {
  __shared__ __align__(16) _Float16 plds_all[8][16 * 32];
  int bid = blockIdx.x;
  int nh = bid / 5;
  int qblk = bid - nh * 5;
  int lane = threadIdx.x & 31;
  int wv = threadIdx.x >> 5;
  int qbase = qblk * 128 + wv * 16;
  if (qbase >= LTOK) return;                        // wave-uniform exit (EXEC stays full)

  const size_t HS = (size_t)LTOK * HD;
  const _Float16* Q  = qkv + (size_t)nh * HS;                       // which = 0
  const _Float16* Kp = qkv + ((size_t)NB * NHD + nh) * HS;          // which = 1
  const _Float16* VT = vt + (size_t)nh * HD * LPAD;
  _Float16* plds = plds_all[wv];

  v16h qa0 = frag_rm_clamp(Q, qbase, LTOK - 1, HD, 0, lane);
  v16h qa1 = frag_rm_clamp(Q, qbase, LTOK - 1, HD, 32, lane);

  v8f acc[4] = {};
  float mrun[8], lrun[8];
#pragma unroll
  for (int r = 0; r < 8; ++r) { mrun[r] = -1e30f; lrun[r] = 0.f; }

  for (int kb = 0; kb < LTOK; kb += 32) {
    // S = Q @ K^T over hd=64 (two k-steps), two 16-key column tiles
    v16h k00 = frag_rm_clamp(Kp, kb,      LTOK - 1, HD, 0,  lane);
    v16h k01 = frag_rm_clamp(Kp, kb,      LTOK - 1, HD, 32, lane);
    v16h k10 = frag_rm_clamp(Kp, kb + 16, LTOK - 1, HD, 0,  lane);
    v16h k11 = frag_rm_clamp(Kp, kb + 16, LTOK - 1, HD, 32, lane);
    v8f s0 = {}, s1 = {};
    s0 = wmma_f16(qa0, k00, s0);
    s0 = wmma_f16(qa1, k01, s0);
    s1 = wmma_f16(qa0, k10, s1);
    s1 = wmma_f16(qa1, k11, s1);

    bool ok0 = (kb + (lane & 15)) < LTOK;
    bool ok1 = (kb + 16 + (lane & 15)) < LTOK;
    float p0[8], p1[8];
#pragma unroll
    for (int r = 0; r < 8; ++r) {
      p0[r] = ok0 ? s0[r] * ATT_SCALE : -1e30f;
      p1[r] = ok1 ? s1[r] * ATT_SCALE : -1e30f;
    }

#pragma unroll
    for (int r = 0; r < 8; ++r) {
      // row max across the 16 lanes of this half (C/D layout: col = lane&15)
      float mx = fmaxf(p0[r], p1[r]);
      mx = fmaxf(mx, __shfl_xor(mx, 1, 16));
      mx = fmaxf(mx, __shfl_xor(mx, 2, 16));
      mx = fmaxf(mx, __shfl_xor(mx, 4, 16));
      mx = fmaxf(mx, __shfl_xor(mx, 8, 16));
      float mnew = fmaxf(mrun[r], mx);
      float alpha = __expf(mrun[r] - mnew);
      mrun[r] = mnew;
      p0[r] = __expf(p0[r] - mnew);
      p1[r] = __expf(p1[r] - mnew);
      float rs = p0[r] + p1[r];
      rs += __shfl_xor(rs, 1, 16);
      rs += __shfl_xor(rs, 2, 16);
      rs += __shfl_xor(rs, 4, 16);
      rs += __shfl_xor(rs, 8, 16);
      lrun[r] = lrun[r] * alpha + rs;
#pragma unroll
      for (int t = 0; t < 4; ++t) acc[t][r] *= alpha;
      // stage P (f16) into LDS in A-fragment row-major [16 rows][32 keys]
      int myrow = r + ((lane >> 4) << 3);
      plds[myrow * 32 + (lane & 15)] = (_Float16)p0[r];
      plds[myrow * 32 + 16 + (lane & 15)] = (_Float16)p1[r];
    }

    // O += P(16x32) @ V(32x64): B fragments contiguous via pre-transposed Vt
    v16h pa = frag_rm(plds, 32, lane);
#pragma unroll
    for (int t = 0; t < 4; ++t) {
      v16h vf = frag_rm(VT + (size_t)t * 16 * LPAD + kb, LPAD, lane);
      acc[t] = wmma_f16(pa, vf, acc[t]);
    }
  }

  int n = nh / NHD, h = nh - n * NHD;
#pragma unroll
  for (int r = 0; r < 8; ++r) {
    int qrow = qbase + r + ((lane >> 4) << 3);
    if (qrow >= LTOK) continue;
    float inv = 1.0f / lrun[r];
    size_t trow = (size_t)n * LTOK + qrow;
#pragma unroll
    for (int t = 0; t < 4; ++t) {
      int col = h * HD + t * 16 + (lane & 15);
      ob[trow * CDIM + col] = (_Float16)(acc[t][r] * inv);
    }
  }
}

// ---------------------------------------------------------------------------
// Depthwise 3x3 conv on V patch tokens, accumulated into the attention output
// ---------------------------------------------------------------------------
__global__ void dwc_kernel(const _Float16* __restrict__ qkv, const float* __restrict__ wd,
                           const float* __restrict__ bd, _Float16* __restrict__ ob) {
  int idx = blockIdx.x * blockDim.x + threadIdx.x;
  const int total = NB * 576 * CDIM;
  if (idx >= total) return;
  int c = idx % CDIM;
  int p = (idx / CDIM) % 576;
  int n = idx / (CDIM * 576);
  int y = p / 24, x = p - y * 24;
  int head = c >> 6, d = c & 63;
  const _Float16* V = qkv + (((size_t)2 * NB * NHD) + (size_t)n * NHD + head) * LTOK * HD;
  float a = bd[c];
#pragma unroll
  for (int ky = -1; ky <= 1; ++ky)
#pragma unroll
    for (int kx = -1; kx <= 1; ++kx) {
      int yy = y + ky, xx = x + kx;
      if (yy < 0 || yy >= 24 || xx < 0 || xx >= 24) continue;
      a += (float)V[(size_t)(1 + yy * 24 + xx) * HD + d] * wd[c * 9 + (ky + 1) * 3 + (kx + 1)];
    }
  size_t o = ((size_t)n * LTOK + 1 + p) * CDIM + c;
  ob[o] = (_Float16)((float)ob[o] + a);
}

// ---------------------------------------------------------------------------
// Projection GEMM: [MTOT x 768] @ [768 x 768]^T + bias -> f32 output
// ---------------------------------------------------------------------------
__global__ __launch_bounds__(256)
void proj_gemm_kernel(const _Float16* __restrict__ ih, const _Float16* __restrict__ wh,
                      const float* __restrict__ bias, float* __restrict__ out) {
  __shared__ __align__(16) _Float16 bsmem[2][64 * 32];
  const int lane = threadIdx.x & 31;
  const int wv = threadIdx.x >> 5;
  int nt = blockIdx.x / MBLKS;
  int mb = blockIdx.x - nt * MBLKS;
  int mt = mb * 8 + wv;
  bool mvalid = mt < MTILES;
  int mtc = mvalid ? mt : MTILES - 1;

  const _Float16* a0 = ih + (size_t)mtc * 16 * CDIM;
  const _Float16* b0 = wh + (size_t)nt * 64 * CDIM;

  v8f acc[4] = {};
  gemm_mainloop(a0, b0, &bsmem[0][0], threadIdx.x, lane, acc);
  if (!mvalid) return;

#pragma unroll
  for (int t = 0; t < 4; ++t) {
    int col = nt * 64 + t * 16 + (lane & 15);
    float b = bias[col];
#pragma unroll
    for (int r = 0; r < 8; ++r) {
      int row = mt * 16 + r + ((lane >> 4) << 3);
      out[(size_t)row * CDIM + col] = acc[t][r] + b;
    }
  }
}

// ---------------------------------------------------------------------------
// Launch pipeline
// ---------------------------------------------------------------------------
extern "C" void kernel_launch(void* const* d_in, const int* in_sizes, int n_in,
                              void* d_out, int out_size, void* d_ws, size_t ws_size,
                              hipStream_t stream) {
  const float* x     = (const float*)d_in[0];
  const float* wqkv  = (const float*)d_in[1];
  const float* wproj = (const float*)d_in[2];
  const float* bproj = (const float*)d_in[3];
  const float* wdwc  = (const float*)d_in[4];
  const float* bdwc  = (const float*)d_in[5];
  float* out = (float*)d_out;

  char* w = (char*)d_ws;
  _Float16* xh     = (_Float16*)w; w += (size_t)MTOT * CDIM * 2;
  _Float16* wqkvh  = (_Float16*)w; w += (size_t)C3 * CDIM * 2;
  _Float16* wprojh = (_Float16*)w; w += (size_t)CDIM * CDIM * 2;
  _Float16* qkvh   = (_Float16*)w; w += (size_t)3 * NB * NHD * LTOK * HD * 2;
  _Float16* vth    = (_Float16*)w; w += (size_t)NB * NHD * HD * LPAD * 2;
  _Float16* ob     = (_Float16*)w; w += (size_t)MTOT * CDIM * 2;

  {
    size_t n = (size_t)MTOT * CDIM;
    f32_to_f16_kernel<<<(unsigned)((n + 255) / 256), 256, 0, stream>>>(x, xh, n);
  }
  {
    size_t n = (size_t)C3 * CDIM;
    f32_to_f16_kernel<<<(unsigned)((n + 255) / 256), 256, 0, stream>>>(wqkv, wqkvh, n);
  }
  {
    size_t n = (size_t)CDIM * CDIM;
    f32_to_f16_kernel<<<(unsigned)((n + 255) / 256), 256, 0, stream>>>(wproj, wprojh, n);
  }

  qkv_gemm_kernel<<<(C3 / 64) * MBLKS, 256, 0, stream>>>(xh, wqkvh, qkvh);
  {
    size_t n = (size_t)NB * NHD * HD * LPAD;
    vt_kernel<<<(unsigned)((n + 255) / 256), 256, 0, stream>>>(qkvh, vth);
  }
  attn_kernel<<<NB * NHD * 5, 256, 0, stream>>>(qkvh, vth, ob);
  {
    int total = NB * 576 * CDIM;
    dwc_kernel<<<(total + 255) / 256, 256, 0, stream>>>(qkvh, wdwc, bdwc, ob);
  }
  proj_gemm_kernel<<<(CDIM / 64) * MBLKS, 256, 0, stream>>>(ob, wprojh, bproj, out);
}
